// Lfm2MoeSparseMoeBlock_4982162063769
// MI455X (gfx1250) — matrix operand order repaired and checked by reading, not compiled, not run on hardware
//
#include <hip/hip_runtime.h>

// Problem constants (match reference)
#define E_  32
#define K_  4
#define H_  2048
#define I_  1024
#define T_  4096
#define MAX_TILES 256   // T_/16 : worst case one expert gets every token

typedef __attribute__((ext_vector_type(16))) __bf16 v16bf;
typedef __attribute__((ext_vector_type(8)))  __bf16 v8bf;
typedef __attribute__((ext_vector_type(8)))  float  v8f;

// ---- helpers -------------------------------------------------------------

__device__ __forceinline__ float fast_sigmoid(float g) {
  // v_exp_f32 + v_rcp_f32 (no IEEE div chain)
  return __builtin_amdgcn_rcpf(1.0f + __expf(-g));
}

// A operand, contiguous K order (matches B):
// lanes 0-15: M=lane, halves 0-15 = K(0..15)
// lanes 16-31 (koff=8): halves 0-15 = K(16..31).
__device__ __forceinline__ v16bf load_a16(const __bf16* rowp, int koff) {
  union { v16bf v; v8bf h[2]; } u;
  u.h[0] = *(const v8bf*)(rowp + 2 * koff);
  u.h[1] = *(const v8bf*)(rowp + 2 * koff + 8);
  return u.v;
}

// B operand: 16 contiguous K values per lane (this lane's half selected by
// kblk added by the caller). Overloads: pre-converted bf16 (fast path) or
// fp32 with on-the-fly conversion (fallback).
__device__ __forceinline__ v16bf load_b(const __bf16* p) {
  union { v16bf v; v8bf h[2]; } u;
  u.h[0] = *(const v8bf*)(p);
  u.h[1] = *(const v8bf*)(p + 8);
  return u.v;
}

__device__ __forceinline__ v16bf load_b(const float* p) {
  union { v16bf v; __bf16 b[16]; } u;
#pragma unroll
  for (int j = 0; j < 16; j += 4) {
    float4 f = *(const float4*)(p + j);
    u.b[j + 0] = (__bf16)f.x; u.b[j + 1] = (__bf16)f.y;
    u.b[j + 2] = (__bf16)f.z; u.b[j + 3] = (__bf16)f.w;
  }
  return u.v;
}

__device__ __forceinline__ v8f wmma_bf16(v16bf a, v16bf b, v8f c) {
  return __builtin_amdgcn_wmma_f32_16x16x32_bf16(
      /*neg_a=*/false, a, /*neg_b=*/false, b,
      /*c_mod=*/(short)0, c, /*reuse_a=*/false, /*reuse_b=*/false);
}

// ---- kernel 0: fp32 -> bf16 weight pre-conversion ------------------------

__global__ __launch_bounds__(256)
void convert_w_kernel(const float* __restrict__ src, __bf16* __restrict__ dst,
                      size_t n4) {             // n4 = element count / 4
  size_t i = (size_t)blockIdx.x * blockDim.x + threadIdx.x;
  size_t stride = (size_t)gridDim.x * blockDim.x;
  for (; i < n4; i += stride) {
    float4 f = *(const float4*)(src + i * 4);
    union { __bf16 b[4]; unsigned long long u; } p;
    p.b[0] = (__bf16)f.x; p.b[1] = (__bf16)f.y;
    p.b[2] = (__bf16)f.z; p.b[3] = (__bf16)f.w;
    *(unsigned long long*)(dst + i * 4) = p.u;
  }
}

// ---- kernel 1: zero output & counters ------------------------------------

__global__ void init_kernel(float* __restrict__ out, int* __restrict__ counts,
                            size_t n) {
  size_t i = (size_t)blockIdx.x * blockDim.x + threadIdx.x;
  size_t stride = (size_t)gridDim.x * blockDim.x;
  for (; i < n; i += stride) out[i] = 0.0f;
  if (blockIdx.x == 0 && threadIdx.x < E_) counts[threadIdx.x] = 0;
}

// ---- kernel 2: router + biased top-k + expert list build -----------------
// One wave32 per token, lane == expert.

__global__ __launch_bounds__(256)
void router_topk_kernel(const float* __restrict__ x,
                        const float* __restrict__ gate_w,
                        const float* __restrict__ bias,
                        int* __restrict__ counts,
                        int* __restrict__ token_ids,
                        float* __restrict__ gate_wt) {
  const int wave = threadIdx.x >> 5;
  const int lane = threadIdx.x & 31;
  const int t = blockIdx.x * 8 + wave;
  if (t >= T_) return;

  const float* xrow = x + (size_t)t * H_;
  const float* gw   = gate_w + (size_t)lane * H_;
  float acc = 0.0f;
  for (int h = 0; h < H_; h += 4) {
    float4 xv = *(const float4*)(xrow + h);
    float4 gv = *(const float4*)(gw + h);
    acc += xv.x * gv.x + xv.y * gv.y + xv.z * gv.z + xv.w * gv.w;
  }
  const float score  = fast_sigmoid(acc);
  float       biased = score + bias[lane];             // selection only

  float my_w = 0.0f;
  int   my_e = 0;
  float wsum = 0.0f;
#pragma unroll
  for (int k = 0; k < K_; ++k) {
    float v = biased;
    int   l = lane;
#pragma unroll
    for (int off = 16; off > 0; off >>= 1) {
      float ov = __shfl_xor(v, off);
      int   ol = __shfl_xor(l, off);
      if (ov > v || (ov == v && ol < l)) { v = ov; l = ol; }
    }
    float sk = __shfl(score, l);     // bias-free weight of the winner
    wsum += sk;
    if (lane == k) { my_e = l; my_w = sk; }
    if (lane == l) biased = -__builtin_inff();
  }

  if (lane < K_) {
    int slot = atomicAdd(&counts[my_e], 1);
    token_ids[my_e * T_ + slot] = t;
    gate_wt[my_e * T_ + slot]   = my_w * __builtin_amdgcn_rcpf(wsum);
  }
}

// ---- kernel 3: fused expert FFN (GEMM1 -> SwiGLU -> GEMM2 -> combine) ----
// grid.x = E_ * MAX_TILES ; block = 256 (8 waves). Each block owns a
// 16-token tile of one expert's token list; intermediates live in LDS.
// BT = __bf16 (pre-converted weights, fast path) or float (on-the-fly cvt).

template <typename BT>
__global__ __launch_bounds__(256)
void expert_kernel(const float* __restrict__ x,
                   const BT*    __restrict__ w13,
                   const BT*    __restrict__ w2,
                   const int*   __restrict__ counts,
                   const int*   __restrict__ token_ids,
                   const float* __restrict__ gate_wt,
                   float* __restrict__ out) {
  __shared__ __bf16 x_lds[16 * H_];   // 64 KB: gathered x tile
  __shared__ __bf16 h_lds[16 * I_];   // 32 KB: SwiGLU activations
  __shared__ int   tok_lds[16];
  __shared__ float cw_lds[16];

  const int e    = blockIdx.x >> 8;        // / MAX_TILES
  const int tile = blockIdx.x & (MAX_TILES - 1);
  const int cnt  = counts[e];
  const int base = tile * 16;
  if (base >= cnt) return;

  const int tid  = threadIdx.x;
  const int wave = tid >> 5;
  const int lane = tid & 31;
  const int n    = lane & 15;                     // column / A-row index
  const int koff = (lane < 16) ? 0 : 8;           // A-operand K sub-offset
  const int kblk = (lane < 16) ? 0 : 16;          // B-operand K half
  const int rbse = (lane < 16) ? 0 : 8;           // C/D row base

  if (tid < 16) {
    int s = base + tid;
    if (s < cnt) {
      tok_lds[tid] = token_ids[e * T_ + s];
      cw_lds[tid]  = gate_wt[e * T_ + s];
    } else {                                      // pad row: weight 0
      tok_lds[tid] = token_ids[e * T_ + base];
      cw_lds[tid]  = 0.0f;
    }
  }
  __syncthreads();

  // gather 16 token rows of x into LDS as bf16
  for (int v = tid; v < 16 * (H_ / 4); v += 256) {
    int row = v / (H_ / 4);
    int c4  = (v % (H_ / 4)) * 4;
    float4 f = *(const float4*)(x + (size_t)tok_lds[row] * H_ + c4);
    union { __bf16 b[4]; unsigned long long u; } p;
    p.b[0] = (__bf16)f.x; p.b[1] = (__bf16)f.y;
    p.b[2] = (__bf16)f.z; p.b[3] = (__bf16)f.w;
    *(unsigned long long*)&x_lds[row * H_ + c4] = p.u;
  }
  __syncthreads();

  // ---- phase 1: h = silu(x @ w13g^T) * (x @ w13u^T)  (per-wave 16x16 tiles)
  const BT* w13e = w13 + (size_t)e * (2 * I_) * H_;
  for (int p = 0; p < 8; ++p) {                        // 8 waves*16 = 128 cols
    const int col0 = p * 128 + wave * 16;
    v8f accg = {};
    v8f accu = {};
    const BT* bg = w13e + (size_t)(col0 + n) * H_ + kblk;        // gate rows
    const BT* bu = w13e + (size_t)(I_ + col0 + n) * H_ + kblk;   // up rows
    for (int kb = 0; kb < H_; kb += 32) {
      __builtin_prefetch(bg + kb + 256, 0, 1);         // global_prefetch_b8
      v16bf a  = load_a16(&x_lds[n * H_ + kb], koff);
      v16bf vg = load_b(bg + kb);
      v16bf vu = load_b(bu + kb);
      accg = wmma_bf16(a, vg, accg);
      accu = wmma_bf16(a, vu, accu);
    }
#pragma unroll
    for (int r = 0; r < 8; ++r) {                      // SwiGLU, bf16 into LDS
      int   row = r + rbse;
      float g   = accg[r];
      float hv  = g * fast_sigmoid(g) * accu[r];       // silu(g) * u
      h_lds[row * I_ + col0 + n] = (__bf16)hv;
    }
  }
  __syncthreads();

  // ---- phase 2: out += cw * (h @ w2^T)
  const BT* w2e = w2 + (size_t)e * H_ * I_;
  for (int p = 0; p < 16; ++p) {                       // 2048 / 128 = 16 passes
    const int col0 = p * 128 + wave * 16;
    v8f acc = {};
    const BT* bb = w2e + (size_t)(col0 + n) * I_ + kblk;
    for (int kb = 0; kb < I_; kb += 32) {
      __builtin_prefetch(bb + kb + 256, 0, 1);
      v16bf a = load_a16(&h_lds[n * I_ + kb], koff);
      v16bf b = load_b(bb + kb);
      acc = wmma_bf16(a, b, acc);
    }
#pragma unroll
    for (int r = 0; r < 8; ++r) {
      int row = r + rbse;
      float v = cw_lds[row] * acc[r];                  // ROUTED_SCALING_FACTOR=1
      atomicAdd(&out[(size_t)tok_lds[row] * H_ + col0 + n], v);
    }
  }
}

// ---- launcher ------------------------------------------------------------

extern "C" void kernel_launch(void* const* d_in, const int* in_sizes, int n_in,
                              void* d_out, int out_size, void* d_ws, size_t ws_size,
                              hipStream_t stream) {
  (void)in_sizes; (void)n_in; (void)out_size;
  const float* x      = (const float*)d_in[0];   // [T,H]
  const float* gate_w = (const float*)d_in[1];   // [E,H]
  const float* bias   = (const float*)d_in[2];   // [E]
  const float* w13    = (const float*)d_in[3];   // [E,2I,H]
  const float* w2     = (const float*)d_in[4];   // [E,H,I]
  float*       out    = (float*)d_out;           // [T,H]

  const size_t n13 = (size_t)E_ * 2 * I_ * H_;   // 134,217,728 elems
  const size_t n2  = (size_t)E_ * H_ * I_;       //  67,108,864 elems

  // workspace layout
  char*   ws        = (char*)d_ws;
  int*    counts    = (int*)ws;                                  // E ints
  int*    token_ids = (int*)(ws + 128);                          // E*T ints
  float*  gate_wt   = (float*)(ws + 128 + (size_t)E_ * T_ * 4);  // E*T floats
  char*   ws_w      = ws + 128 + (size_t)E_ * T_ * 8;            // bf16 weights
  __bf16* w13b      = (__bf16*)ws_w;
  __bf16* w2b       = (__bf16*)(ws_w + n13 * 2);
  const size_t need = (size_t)(ws_w - ws) + (n13 + n2) * 2;

  init_kernel<<<1024, 256, 0, stream>>>(out, counts, (size_t)T_ * H_);
  router_topk_kernel<<<T_ / 8, 256, 0, stream>>>(x, gate_w, bias, counts,
                                                 token_ids, gate_wt);
  if (ws_size >= need) {
    // fast path: pre-convert weights to bf16, GEMMs stream bf16 directly
    convert_w_kernel<<<2048, 256, 0, stream>>>(w13, w13b, n13 / 4);
    convert_w_kernel<<<2048, 256, 0, stream>>>(w2, w2b, n2 / 4);
    expert_kernel<__bf16><<<E_ * MAX_TILES, 256, 0, stream>>>(
        x, w13b, w2b, counts, token_ids, gate_wt, out);
  } else {
    // fallback: convert weights on the fly inside the K-loop
    expert_kernel<float><<<E_ * MAX_TILES, 256, 0, stream>>>(
        x, w13, w2, counts, token_ids, gate_wt, out);
  }
}
